// TransformerEncoderLayer_69483980914753
// MI455X (gfx1250) — matrix operand order repaired and checked by reading, not compiled, not run on hardware
//
#include <hip/hip_runtime.h>
#include <hip/hip_bf16.h>

// Problem constants (B,L,D,H from the reference)
#define B_  2
#define L_  256
#define D_  512
#define H_  8
#define DH_ 64
#define BL_ (B_*L_)
#define NEGF (-1000000000000000.0f)

typedef __bf16 bf16;
typedef bf16  v16bf __attribute__((ext_vector_type(16)));
typedef float v8f   __attribute__((ext_vector_type(8)));

// ---------------------------------------------------------------------------
// WMMA layouts used here (wave32, v_wmma_f32_16x16x32_bf16; A/B k order is this file's, not AMD's table):
//  A 16x32: lane m=lane&15, hi=lane>>4; half p = A[m][K=hi*16+p]
//  B 32x16: lane n=lane&15, hi=lane>>4; half p  = B[K=hi*16+p][n]
//  C/D 16x16: VGPR r holds M = r + 8*(lane>=16), N = lane&15
//
// Swizzled B storage: for logical B[K x N], fragment tile (kt,nt) is 512
// contiguous halves: [lane 0..31][16 halves], lane = (k>>4 & 1)*16 + (n&15),
// half t = k&15.  A B-fragment load is then one aligned 32-byte read per lane.
// ---------------------------------------------------------------------------
__device__ __forceinline__ v8f zero8() {
  v8f z;
#pragma unroll
  for (int i = 0; i < 8; ++i) z[i] = 0.0f;
  return z;
}

__device__ __forceinline__ v8f wmma_bf16(v16bf a, v16bf b, v8f c) {
  return __builtin_amdgcn_wmma_f32_16x16x32_bf16(false, a, false, b,
                                                 (short)0, c, false, false);
}

// A fragment from row-major bf16 [16 x 32] tile at src, leading dim lda
__device__ __forceinline__ v16bf load_a_frag(const bf16* src, int lda) {
  int lane = threadIdx.x & 31;
  int m = lane & 15, hi = lane >> 4;
  const bf16* p = src + (long)m * lda;
  v16bf a;
#pragma unroll
  for (int t = 0; t < 8; ++t) { a[t] = p[hi * 16 + t]; a[8 + t] = p[hi * 16 + 8 + t]; }
  return a;
}

// Same, but only rows m<8 are valid (rows 8..15 are zero padding: H=8 heads)
__device__ __forceinline__ v16bf load_a_frag_pad8(const bf16* src, int lda) {
  int lane = threadIdx.x & 31;
  int m = lane & 15, hi = lane >> 4;
  v16bf a;
  if (m < 8) {
    const bf16* p = src + (long)m * lda;
#pragma unroll
    for (int t = 0; t < 8; ++t) { a[t] = p[hi * 16 + t]; a[8 + t] = p[hi * 16 + 8 + t]; }
  } else {
#pragma unroll
    for (int t = 0; t < 16; ++t) a[t] = (bf16)0.0f;
  }
  return a;
}

// B fragment from swizzled storage: one aligned 32B vector load per lane.
__device__ __forceinline__ v16bf load_b_swz(const bf16* base, int kt, int nt,
                                            int ntiles_n) {
  int lane = threadIdx.x & 31;
  return *(const v16bf*)(base + (((long)kt * ntiles_n + nt) * 32 + lane) * 16);
}

// B fragment where B[k][n] = src[n*lds + k] (f32 source, transposed access):
// per lane the 16 K-values are CONTIGUOUS floats -> b128 loads + cvt to bf16.
__device__ __forceinline__ v16bf load_b_frag_f32T(const float* src, int lds) {
  int lane = threadIdx.x & 31;
  int n = lane & 15, hi = lane >> 4;
  const float* p = src + (long)n * lds + hi * 16;
  v16bf b;
#pragma unroll
  for (int t = 0; t < 16; ++t) b[t] = (bf16)p[t];
  return b;
}

__device__ __forceinline__ void store_d_f32(float* dst, int ldd, v8f acc) {
  int lane = threadIdx.x & 31;
  int n = lane & 15, hi = lane >> 4;
#pragma unroll
  for (int r = 0; r < 8; ++r) dst[(long)(r + 8 * hi) * ldd + n] = acc[r];
}

__device__ __forceinline__ float wave_sum(float x) {
#pragma unroll
  for (int off = 16; off > 0; off >>= 1) x += __shfl_xor(x, off, 32);
  return x;
}
__device__ __forceinline__ float wave_max(float x) {
#pragma unroll
  for (int off = 16; off > 0; off >>= 1) x = fmaxf(x, __shfl_xor(x, off, 32));
  return x;
}

// ---------------------------------------------------------------------------
// K0: inputs->bf16 copy; Wq/Wk/Wv/Wf transposed+swizzled; Wr swizzled.
// For swizzled outputs, tid decomposes as:
//   t = tid&15 (K low), lane = (tid>>4)&31, tile = tid>>9, nt = tile%32, kt = tile/32
//   k = kt*32 + (lane>>4)*16 + t,  n = nt*16 + (lane&15)
// Transposed weights: B[k][n] = W[n][k]  (k fast-varying -> coalesced reads).
// grid (1024, 6) x 256
// ---------------------------------------------------------------------------
__global__ void prep_kernel(const float* __restrict__ inputs,
                            const float* __restrict__ Wq, const float* __restrict__ Wk,
                            const float* __restrict__ Wv, const float* __restrict__ Wf,
                            const float* __restrict__ Wr,
                            bf16* in_bf, bf16* wq_swz, bf16* wk_swz, bf16* wv_swz,
                            bf16* wf_swz, bf16* wr_swz) {
  int tid = blockIdx.x * 256 + threadIdx.x;   // 0 .. 262143
  int t = tid & 15;
  int lane = (tid >> 4) & 31;
  int tile = tid >> 9;
  int nt = tile & 31, kt = tile >> 5;
  int k = kt * 32 + (lane >> 4) * 16 + t;
  int n = nt * 16 + (lane & 15);
  switch (blockIdx.y) {
    case 0: in_bf[tid]  = (bf16)inputs[tid]; break;
    case 1: wq_swz[tid] = (bf16)Wq[(long)n * D_ + k]; break;
    case 2: wk_swz[tid] = (bf16)Wk[(long)n * D_ + k]; break;
    case 3: wv_swz[tid] = (bf16)Wv[(long)n * D_ + k]; break;
    case 4: wf_swz[tid] = (bf16)Wf[(long)n * D_ + k]; break;
    case 5: wr_swz[tid] = (bf16)Wr[(long)k * D_ + n]; break;  // no transpose
  }
}

// ---------------------------------------------------------------------------
// K1: fused QKV GEMM.  [BL x D] @ [D x D] tiles; epilogue writes
//   mat 0: qu = q+bq+u and qpv = q+bq+v  (row-major A layout, [B,H,L,DH])
//   mat 1: k+bk  scattered into swizzled B layout (K=DH, N=L per (b,h))
//   mat 2: v+bv  scattered into swizzled B layout (K=L, N=DH per (b,h))
// grid (1024, 3) x 32
// ---------------------------------------------------------------------------
__global__ void qkv_gemm_kernel(const bf16* __restrict__ in_bf,
                                const bf16* __restrict__ wq_swz,
                                const bf16* __restrict__ wk_swz,
                                const bf16* __restrict__ wv_swz,
                                const float* __restrict__ bq, const float* __restrict__ bk,
                                const float* __restrict__ bv,
                                const float* __restrict__ u, const float* __restrict__ v,
                                bf16* qu, bf16* qpv, bf16* k_swz, bf16* v_swz) {
  int tile = blockIdx.x, mat = blockIdx.y;
  int row0 = (tile >> 5) << 4;    // 32 row tiles over BL=512
  int col0 = (tile & 31) << 4;    // 32 col tiles over D=512
  const bf16* w = (mat == 0) ? wq_swz : (mat == 1) ? wk_swz : wv_swz;
  v8f acc = zero8();
#pragma unroll 4
  for (int kt = 0; kt < 16; ++kt) {
    v16bf a = load_a_frag(in_bf + (long)row0 * D_ + kt * 32, D_);
    v16bf b = load_b_swz(w, kt, col0 >> 4, 32);
    acc = wmma_bf16(a, b, acc);
  }
  int lane = threadIdx.x & 31, n = lane & 15, hi = lane >> 4;
  int dout = col0 + n, h = dout >> 6, dh = dout & 63;
  float bias = ((mat == 0) ? bq : (mat == 1) ? bk : bv)[dout];
#pragma unroll
  for (int r = 0; r < 8; ++r) {
    int row = row0 + r + 8 * hi;
    int bi = row >> 8, il = row & 255;
    float val = acc[r] + bias;
    long bh_ofs = (long)(bi * H_ + h) * 16384;   // per-(b,h) swizzled panel
    if (mat == 0) {
      long o = (((long)(bi * H_ + h) * L_ + il) * DH_) + dh;
      qu[o]  = (bf16)(val + u[h * DH_ + dh]);
      qpv[o] = (bf16)(val + v[h * DH_ + dh]);
    } else if (mat == 1) {
      // B(k=dh, n=il): K=64 (2 kt), N=256 (16 nt)
      long idx = ((long)((dh >> 5) * 16 + (il >> 4)) * 32 +
                  ((dh >> 4) & 1) * 16 + (il & 15)) * 16 + (dh & 15);
      k_swz[bh_ofs + idx] = (bf16)val;
    } else {
      // B(k=il, n=dh): K=256 (8 kt), N=64 (4 nt)
      long idx = ((long)((il >> 5) * 4 + (dh >> 4)) * 32 +
                  ((il >> 4) & 1) * 16 + (dh & 15)) * 16 + (il & 15);
      v_swz[bh_ofs + idx] = (bf16)val;
    }
  }
}

// ---------------------------------------------------------------------------
// K2: qvproj[b,i,h,:] = qpv[b,h,i,:] @ Wr[h*64:(h+1)*64, :]   ([L x 64]@[64 x D])
// Wr swizzled with global K=512: head h uses k-tiles (h*2 + kt).
// grid (512, 16) x 32
// ---------------------------------------------------------------------------
__global__ void qvproj_gemm_kernel(const bf16* __restrict__ qpv,
                                   const bf16* __restrict__ wr_swz,
                                   bf16* qvproj) {
  int bh = blockIdx.y, b = bh >> 3, h = bh & 7;
  int tile = blockIdx.x;
  int row0 = (tile >> 5) << 4;    // i: 16 tiles
  int col0 = (tile & 31) << 4;    // d: 32 tiles
  v8f acc = zero8();
  const bf16* abase = qpv + ((long)(b * H_ + h) * L_ + row0) * DH_;
#pragma unroll
  for (int kt = 0; kt < 2; ++kt) {
    v16bf a = load_a_frag(abase + kt * 32, DH_);
    v16bf bb = load_b_swz(wr_swz, h * 2 + kt, col0 >> 4, 32);
    acc = wmma_bf16(a, bb, acc);
  }
  int lane = threadIdx.x & 31, n = lane & 15, hi = lane >> 4;
#pragma unroll
  for (int r = 0; r < 8; ++r) {
    int i = row0 + r + 8 * hi;
    qvproj[((long)(b * L_ + i) * H_ + h) * D_ + col0 + n] = (bf16)acc[r];
  }
}

// ---------------------------------------------------------------------------
// K3: A+C term: score[b,h,i,j] = qu[b,h,i,:] . k[b,h,j,:]
// grid (256, 16) x 32
// ---------------------------------------------------------------------------
__global__ void ac_gemm_kernel(const bf16* __restrict__ qu,
                               const bf16* __restrict__ k_swz,
                               float* score) {
  int bh = blockIdx.y;
  int tile = blockIdx.x;
  int row0 = (tile >> 4) << 4, col0 = (tile & 15) << 4;
  v8f acc = zero8();
  const bf16* abase = qu + ((long)bh * L_ + row0) * DH_;
  const bf16* bbase = k_swz + (long)bh * 16384;
#pragma unroll
  for (int kt = 0; kt < 2; ++kt) {
    v16bf a = load_a_frag(abase + kt * 32, DH_);
    v16bf bb = load_b_swz(bbase, kt, col0 >> 4, 16);
    acc = wmma_bf16(a, bb, acc);
  }
  store_d_f32(score + ((long)bh * L_ + row0) * L_ + col0, L_, acc);
}

// ---------------------------------------------------------------------------
// K4: B+D term, the bandwidth-dominant kernel (streams pos_embedding ONCE).
// For fixed (b,i): out[h,j] = qvproj[b,i,h,:] . pos[b,i,j,:]   (h padded 8->16)
// Each wave: one (b,i), 4 j-tiles (64 j), K=512, accumulates into score.
// grid (512, 4) x 32
// ---------------------------------------------------------------------------
__global__ void bd_gemm_kernel(const float* __restrict__ pos,
                               const bf16* __restrict__ qvproj,
                               float* score) {
  int bi = blockIdx.x, b = bi >> 8, i = bi & 255;
  int jc = blockIdx.y;
  v8f acc[4];
#pragma unroll
  for (int s = 0; s < 4; ++s) acc[s] = zero8();
  const bf16* abase = qvproj + (long)bi * H_ * D_;
  const float* pbase = pos + (long)bi * L_ * D_;
  for (int kt = 0; kt < 16; ++kt) {
    if (kt < 15)
      __builtin_prefetch(pbase + (long)(jc * 64) * D_ + (kt + 1) * 32, 0, 0);
    v16bf a = load_a_frag_pad8(abase + kt * 32, D_);
#pragma unroll
    for (int s = 0; s < 4; ++s) {
      int j0 = jc * 64 + s * 16;
      v16bf bb = load_b_frag_f32T(pbase + (long)j0 * D_ + kt * 32, D_);
      acc[s] = wmma_bf16(a, bb, acc[s]);
    }
  }
  int lane = threadIdx.x & 31, n = lane & 15, hi = lane >> 4;
  if (hi == 0) {            // rows m=0..7 are the real heads; 8..15 are padding
#pragma unroll
    for (int s = 0; s < 4; ++s) {
      int j = jc * 64 + s * 16 + n;
#pragma unroll
      for (int r = 0; r < 8; ++r) {
        long idx = (((long)(b * H_ + r) * L_ + i) * L_) + j;
        score[idx] += acc[s][r];   // unique (b,h,i,j) per wave -> no race
      }
    }
  }
}

// ---------------------------------------------------------------------------
// K5: scale + key mask + softmax over j, emit bf16 attn (row-major A layout).
// grid (B*H*L) x 32
// ---------------------------------------------------------------------------
__global__ void softmax_kernel(const float* __restrict__ score,
                               const int* __restrict__ seq_mask,
                               bf16* attn) {
  int row = blockIdx.x;          // (b*H+h)*L + i
  int b = row >> 11;             // H*L = 2048
  int lane = threadIdx.x;
  const float* sp = score + (long)row * L_;
  float vals[8], m = -3.4e38f;
#pragma unroll
  for (int t = 0; t < 8; ++t) {
    int j = lane * 8 + t;
    float x = sp[j] * 0.125f;    // 1/sqrt(DH)
    if (seq_mask[b * L_ + j] == 0) x = NEGF;
    vals[t] = x;
    m = fmaxf(m, x);
  }
  m = wave_max(m);
  float sum = 0.0f;
#pragma unroll
  for (int t = 0; t < 8; ++t) { vals[t] = __expf(vals[t] - m); sum += vals[t]; }
  sum = wave_sum(sum);
  float inv = 1.0f / sum;
  bf16* ap = attn + (long)row * L_;
#pragma unroll
  for (int t = 0; t < 8; ++t) ap[lane * 8 + t] = (bf16)(vals[t] * inv);
}

// ---------------------------------------------------------------------------
// K6: ctx[b,i,h*64+dh] = sum_j attn[b,h,i,j] * v[b,h,j,dh]
// grid (64, 16) x 32
// ---------------------------------------------------------------------------
__global__ void ctx_gemm_kernel(const bf16* __restrict__ attn,
                                const bf16* __restrict__ v_swz,
                                bf16* ctx) {
  int bh = blockIdx.y, b = bh >> 3, h = bh & 7;
  int tile = blockIdx.x;
  int row0 = (tile >> 2) << 4, col0 = (tile & 3) << 4;
  v8f acc = zero8();
  const bf16* abase = attn + ((long)bh * L_ + row0) * L_;
  const bf16* bbase = v_swz + (long)bh * 16384;
#pragma unroll
  for (int kt = 0; kt < 8; ++kt) {
    v16bf a = load_a_frag(abase + kt * 32, L_);
    v16bf bb = load_b_swz(bbase, kt, col0 >> 4, 4);
    acc = wmma_bf16(a, bb, acc);
  }
  int lane = threadIdx.x & 31, n = lane & 15, hi = lane >> 4;
#pragma unroll
  for (int r = 0; r < 8; ++r) {
    int i = row0 + r + 8 * hi;
    ctx[((long)(b * L_ + i)) * D_ + h * DH_ + col0 + n] = (bf16)acc[r];
  }
}

// ---------------------------------------------------------------------------
// K7: FFN + residual: ffn = ctx @ Wf^T + bf + inputs   (f32 out)
// grid (1024) x 32
// ---------------------------------------------------------------------------
__global__ void ffn_gemm_kernel(const bf16* __restrict__ ctx,
                                const bf16* __restrict__ wf_swz,
                                const float* __restrict__ bfv,
                                const float* __restrict__ inputs,
                                float* ffn) {
  int tile = blockIdx.x;
  int row0 = (tile >> 5) << 4, col0 = (tile & 31) << 4;
  v8f acc = zero8();
#pragma unroll 4
  for (int kt = 0; kt < 16; ++kt) {
    v16bf a = load_a_frag(ctx + (long)row0 * D_ + kt * 32, D_);
    v16bf bb = load_b_swz(wf_swz, kt, col0 >> 4, 32);
    acc = wmma_bf16(a, bb, acc);
  }
  int lane = threadIdx.x & 31, n = lane & 15, hi = lane >> 4;
  int dout = col0 + n;
  float bias = bfv[dout];
#pragma unroll
  for (int r = 0; r < 8; ++r) {
    int row = row0 + r + 8 * hi;
    ffn[(long)row * D_ + dout] = acc[r] + bias + inputs[(long)row * D_ + dout];
  }
}

// ---------------------------------------------------------------------------
// K8: LayerNorm over last dim. One wave per row (512 elems, 16/lane).
// grid (BL) x 32
// ---------------------------------------------------------------------------
__global__ void layernorm_kernel(const float* __restrict__ ffn,
                                 const float* __restrict__ gamma,
                                 const float* __restrict__ beta,
                                 float* out) {
  int row = blockIdx.x, lane = threadIdx.x;
  const float* xp = ffn + (long)row * D_;
  float x[16], s = 0.0f;
#pragma unroll
  for (int t = 0; t < 16; ++t) { x[t] = xp[lane * 16 + t]; s += x[t]; }
  s = wave_sum(s);
  float mu = s * (1.0f / D_);
  float vs = 0.0f;
#pragma unroll
  for (int t = 0; t < 16; ++t) { float d = x[t] - mu; vs += d * d; }
  vs = wave_sum(vs);
  float rstd = rsqrtf(vs * (1.0f / D_) + 1e-12f);
  float* op = out + (long)row * D_;
#pragma unroll
  for (int t = 0; t < 16; ++t) {
    int c = lane * 16 + t;
    op[c] = (x[t] - mu) * rstd * gamma[c] + beta[c];
  }
}

// ---------------------------------------------------------------------------
extern "C" void kernel_launch(void* const* d_in, const int* in_sizes, int n_in,
                              void* d_out, int out_size, void* d_ws, size_t ws_size,
                              hipStream_t stream) {
  (void)in_sizes; (void)n_in; (void)out_size; (void)ws_size;
  const float* inputs = (const float*)d_in[0];
  const float* pos    = (const float*)d_in[1];
  const int*   seqm   = (const int*)  d_in[2];
  const float* Wq = (const float*)d_in[3];  const float* bq = (const float*)d_in[4];
  const float* Wk = (const float*)d_in[5];  const float* bk = (const float*)d_in[6];
  const float* Wv = (const float*)d_in[7];  const float* bv = (const float*)d_in[8];
  const float* Wr = (const float*)d_in[9];  /* br (d_in[10]) is softmax-shift-invariant */
  const float* u  = (const float*)d_in[11]; const float* v  = (const float*)d_in[12];
  const float* Wf = (const float*)d_in[13]; const float* bfv= (const float*)d_in[14];
  const float* gamma = (const float*)d_in[15];
  const float* beta  = (const float*)d_in[16];
  float* out = (float*)d_out;

  char* ws = (char*)d_ws;
  size_t cursor = 0;
  auto alloc = [&](size_t bytes) -> char* {
    char* p = ws + cursor;
    cursor = (cursor + bytes + 255) & ~(size_t)255;
    return p;
  };
  bf16* in_bf  = (bf16*)alloc((size_t)BL_ * D_ * 2);
  bf16* wq_swz = (bf16*)alloc((size_t)D_ * D_ * 2);
  bf16* wk_swz = (bf16*)alloc((size_t)D_ * D_ * 2);
  bf16* wv_swz = (bf16*)alloc((size_t)D_ * D_ * 2);
  bf16* wf_swz = (bf16*)alloc((size_t)D_ * D_ * 2);
  bf16* wr_swz = (bf16*)alloc((size_t)D_ * D_ * 2);
  bf16* qu     = (bf16*)alloc((size_t)B_ * H_ * L_ * DH_ * 2);
  bf16* qpv    = (bf16*)alloc((size_t)B_ * H_ * L_ * DH_ * 2);
  bf16* k_swz  = (bf16*)alloc((size_t)B_ * H_ * L_ * DH_ * 2);
  bf16* v_swz  = (bf16*)alloc((size_t)B_ * H_ * L_ * DH_ * 2);
  bf16* qvproj = (bf16*)alloc((size_t)B_ * L_ * H_ * D_ * 2);
  float* score = (float*)alloc((size_t)B_ * H_ * L_ * L_ * 4);
  bf16* attn   = (bf16*)alloc((size_t)B_ * H_ * L_ * L_ * 2);
  bf16* ctx    = (bf16*)alloc((size_t)BL_ * D_ * 2);
  float* ffn   = (float*)alloc((size_t)BL_ * D_ * 4);

  prep_kernel<<<dim3(1024, 6), dim3(256), 0, stream>>>(
      inputs, Wq, Wk, Wv, Wf, Wr, in_bf, wq_swz, wk_swz, wv_swz, wf_swz, wr_swz);

  qkv_gemm_kernel<<<dim3(1024, 3), dim3(32), 0, stream>>>(
      in_bf, wq_swz, wk_swz, wv_swz, bq, bk, bv, u, v, qu, qpv, k_swz, v_swz);

  qvproj_gemm_kernel<<<dim3(512, 16), dim3(32), 0, stream>>>(qpv, wr_swz, qvproj);

  ac_gemm_kernel<<<dim3(256, 16), dim3(32), 0, stream>>>(qu, k_swz, score);

  bd_gemm_kernel<<<dim3(512, 4), dim3(32), 0, stream>>>(pos, qvproj, score);

  softmax_kernel<<<dim3(B_ * H_ * L_), dim3(32), 0, stream>>>(score, seqm, attn);

  ctx_gemm_kernel<<<dim3(64, 16), dim3(32), 0, stream>>>(attn, v_swz, ctx);

  ffn_gemm_kernel<<<dim3(1024), dim3(32), 0, stream>>>(ctx, wf_swz, bfv, inputs, ffn);

  layernorm_kernel<<<dim3(BL_), dim3(32), 0, stream>>>(ffn, gamma, beta, out);
}
